// ValueModel_84301618086044
// MI455X (gfx1250) — matrix operand order repaired and checked by reading, not compiled, not run on hardware
//
#include <hip/hip_runtime.h>

typedef __attribute__((ext_vector_type(16))) _Float16 v16h;
typedef __attribute__((ext_vector_type(8)))  float    v8f;
typedef __attribute__((ext_vector_type(4)))  float    v4f;

#define H_MAX 4096

__device__ __constant__ float kFP4[16] = {
    0.0f, 0.0052083333f, 0.6666667f, 1.0f, 0.3333333f, 0.5f, 0.16666667f, 0.25f,
   -0.0f,-0.0052083333f,-0.6666667f,-1.0f,-0.3333333f,-0.5f,-0.16666667f,-0.25f};

// Each wave32 computes 16 rows of the GEMV via v_wmma_f32_16x16x32_f16.
// A = 16x32 tile of x (fp32 streamed NT, converted to f16 in-regs).
// B = w[k0..k0+31] broadcast across all 16 columns (from LDS).
// D column 0 carries the dot products; lane0 holds rows 0-7, lane16 rows 8-15.
__global__ __launch_bounds__(256) void fp4_gemv_wmma(
    const float* __restrict__ x, const int* __restrict__ codes,
    const float* __restrict__ absmax, const float* __restrict__ bias,
    float* __restrict__ out, int N, int H)
{
    __shared__ __align__(32) _Float16 wf16[H_MAX];

    // Dequantize fp4 weight once per block into LDS (8 KB).
    for (int h = threadIdx.x; h < H; h += blockDim.x) {
        int c = codes[h] & 15;
        wf16[h] = (_Float16)(kFP4[c] * absmax[h >> 6]);
    }
    __syncthreads();

    const int lane = threadIdx.x & 31;   // wave32
    const int wave = threadIdx.x >> 5;
    const int hi   = lane >> 4;          // 0: lanes 0-15, 1: lanes 16-31
    const int m    = lane & 15;          // row within the 16-row tile
    const int rowBase = blockIdx.x * 128 + wave * 16;

    int row = rowBase + m;
    if (row >= N) row = N - 1;           // clamp keeps k-loop wave-uniform
    const float* xr = x + (size_t)row * H + 16 * hi;
    const _Float16* wp = &wf16[16 * hi];

    v8f acc = {};
    // Unroll x4 so the compiler can clause 16 global_load_b128s per wait and
    // keep 4 WMMA iterations of loads in flight (MLP >> latency for 23.3TB/s).
#pragma unroll 4
    for (int k0 = 0; k0 < H; k0 += 32) {
        // A-tile per-lane: K = {k0+16*hi .. +15}
        const v4f* p = (const v4f*)(xr + k0);
        v4f a0 = __builtin_nontemporal_load(p);
        v4f a1 = __builtin_nontemporal_load(p + 1);
        v4f a2 = __builtin_nontemporal_load(p + 2);
        v4f a3 = __builtin_nontemporal_load(p + 3);

        v16h a;
        a[0]  = (_Float16)a0[0]; a[1]  = (_Float16)a0[1];
        a[2]  = (_Float16)a0[2]; a[3]  = (_Float16)a0[3];
        a[4]  = (_Float16)a1[0]; a[5]  = (_Float16)a1[1];
        a[6]  = (_Float16)a1[2]; a[7]  = (_Float16)a1[3];
        a[8]  = (_Float16)a2[0]; a[9]  = (_Float16)a2[1];
        a[10] = (_Float16)a2[2]; a[11] = (_Float16)a2[3];
        a[12] = (_Float16)a3[0]; a[13] = (_Float16)a3[1];
        a[14] = (_Float16)a3[2]; a[15] = (_Float16)a3[3];

        // B per-lane: 16 consecutive w halves (lanes 0-15: K=k0.., lanes 16-31: K=k0+16..)
        v16h b = *(const v16h*)(wp + k0);

        acc = __builtin_amdgcn_wmma_f32_16x16x32_f16(
            /*neg_a=*/false, a, /*neg_b=*/false, b,
            /*c_mod=*/(short)0, acc, /*reuse_a=*/false, /*reuse_b=*/false);
    }

    // D-matrix N=0 column: lane 0 -> rows rowBase..+7 (VGPR j = row j),
    //                      lane 16 -> rows rowBase+8..+15.
    if (m == 0) {
        const float bv = bias[0];
        const int ob = rowBase + 8 * hi;
#pragma unroll
        for (int j = 0; j < 8; ++j) {
            int r = ob + j;
            if (r < N) out[r] = acc[j] + bv;
        }
    }
}

extern "C" void kernel_launch(void* const* d_in, const int* in_sizes, int n_in,
                              void* d_out, int out_size, void* d_ws, size_t ws_size,
                              hipStream_t stream) {
    const float* x      = (const float*)d_in[0];
    const int*   codes  = (const int*)  d_in[1];
    const float* absmax = (const float*)d_in[2];
    const float* bias   = (const float*)d_in[3];
    float*       out    = (float*)d_out;

    const int H = in_sizes[1];            // 4096
    const int N = in_sizes[0] / H;        // 32768

    dim3 block(256);                      // 8 wave32 -> 128 rows per block
    dim3 grid((N + 127) / 128);
    fp4_gemv_wmma<<<grid, block, 0, stream>>>(x, codes, absmax, bias, out, N, H);
}
